// ClassAttention_32203664785566
// MI455X (gfx1250) — compile-verified, matrix operand order checked by reading, not run on hardware
//
#include <hip/hip_runtime.h>
#include <hip/hip_bf16.h>
#include <stdint.h>

// Problem constants (from reference)
#define C_DIM   768
#define HEADS   12
#define HDIM    64
#define BATCH   64
#define NTOK    577
#define MROWS   (BATCH * NTOK)   // 36928
#define KV_N    1536
#define SCALE_F 0.125f           // 64^-0.5

typedef __attribute__((ext_vector_type(16))) __bf16 v16bf;
typedef __attribute__((ext_vector_type(8)))  float  v8f;

union FragBF {
  v16bf v;
  uint4 q[2];   // 8 dwords = 16 bf16
};

__device__ __forceinline__ unsigned short f2bf(float f) {
  unsigned int u = __float_as_uint(f);
  u += 0x7FFFu + ((u >> 16) & 1u);         // round-to-nearest-even
  return (unsigned short)(u >> 16);
}
__device__ __forceinline__ float bf2f(unsigned short h) {
  return __uint_as_float(((unsigned int)h) << 16);
}

__device__ __forceinline__ void store_out(float* p, float v)          { *p = v; }
__device__ __forceinline__ void store_out(unsigned short* p, float v) { *p = f2bf(v); }

// ---------------------------------------------------------------------------
// CDNA5 async global->LDS copy (ISA 15.18.3 opcode 98, tracked by ASYNCcnt).
// VDST = per-lane LDS byte address; for a generic __shared__ pointer the LDS
// byte offset is addr[31:0] (flat LDS-aperture mapping, ISA 10.2).
// ---------------------------------------------------------------------------
__device__ __forceinline__ uint32_t lds_off(const void* p) {
  return (uint32_t)(uintptr_t)p;
}
__device__ __forceinline__ void async_load_b128(uint32_t ldsaddr, const void* gaddr) {
  asm volatile("global_load_async_to_lds_b128 %0, %1, off"
               :: "v"(ldsaddr), "v"((uint64_t)(uintptr_t)gaddr)
               : "memory");
}
__device__ __forceinline__ void wait_asynccnt0() {
  asm volatile("s_wait_asynccnt 0" ::: "memory");
}

// ---------------------------------------------------------------------------
// fp32 -> bf16 bulk convert (grid-stride)
// ---------------------------------------------------------------------------
__global__ void f32_to_bf16_kernel(const float* __restrict__ in,
                                   unsigned short* __restrict__ out, int n) {
  for (int i = blockIdx.x * blockDim.x + threadIdx.x; i < n;
       i += gridDim.x * blockDim.x)
    out[i] = f2bf(in[i]);
}

// Gather CLS rows x[b,0,:] -> compact [64,768] bf16
__global__ void gather_cls_kernel(const float* __restrict__ x,
                                  unsigned short* __restrict__ out) {
  int i = blockIdx.x * blockDim.x + threadIdx.x;          // < 64*768
  if (i >= BATCH * C_DIM) return;
  int b = i / C_DIM, c = i % C_DIM;
  out[i] = f2bf(x[(size_t)b * NTOK * C_DIM + c]);
}

// ---------------------------------------------------------------------------
// NT GEMM: C[m,n] = sum_k A[m,k]*B[n,k] + bias[n]
// A: [M,K] bf16 row-major (lda=K), B: [N,K] bf16 row-major (ldb=K).
// Block tile 128x128, 8 waves (4 in M x 2 in N), wave tile 32x64,
// K-step 32 via v_wmma_f32_16x16x32_bf16, fp32 accumulation.
// Double-buffered LDS fed by GLOBAL_LOAD_ASYNC_TO_LDS_B128: tile s+1 streams
// into buf^1 while the 8 WMMAs of tile s execute; s_wait_asynccnt 0 + the
// workgroup barrier at the loop top order producer vs consumers.
// N, K assumed multiples of 128/32; M handled by clamped loads + guarded
// stores (out-of-range A rows only influence never-stored D rows).
// ---------------------------------------------------------------------------
#define TLDS 20   // dword row stride: 16 used + 4 pad (16B-aligned rows, conflict-free)

template <typename OutT>
__global__ __launch_bounds__(256)
void gemm_bf16_nt(const unsigned short* __restrict__ A,
                  const unsigned short* __restrict__ Bm,
                  const float* __restrict__ bias,
                  OutT* __restrict__ C, long long ldc,
                  int M, int N, int K)
{
  __shared__ unsigned int lx[2][128 * TLDS];
  __shared__ unsigned int lw[2][128 * TLDS];

  const int tid  = threadIdx.x;
  const int lane = tid & 31;
  const int wave = tid >> 5;
  const int wm   = wave & 3;   // 0..3 -> M quadrant (32 rows)
  const int wn   = wave >> 2;  // 0..1 -> N half (64 cols)
  const int bm   = blockIdx.y * 128;
  const int bn   = blockIdx.x * 128;
  const int half = lane >> 4;  // 0/1
  const int rsel = lane & 15;

  // Cooperative-load coordinates: 128 rows x 32 bf16 = 512 uint4 per tile,
  // each thread owns rows (tid>>2) and (tid>>2)+64 at uint4 column (tid&3).
  const int lrow0 = tid >> 2;          // 0..63
  const int lrow1 = lrow0 + 64;        // 64..127
  const int c4    = tid & 3;
  const int rA0 = min(bm + lrow0, M - 1);
  const int rA1 = min(bm + lrow1, M - 1);
  const int rB0 = min(bn + lrow0, N - 1);
  const int rB1 = min(bn + lrow1, N - 1);
  const uint4* Aq0 = reinterpret_cast<const uint4*>(A  + (size_t)rA0 * K) + c4;
  const uint4* Aq1 = reinterpret_cast<const uint4*>(A  + (size_t)rA1 * K) + c4;
  const uint4* Bq0 = reinterpret_cast<const uint4*>(Bm + (size_t)rB0 * K) + c4;
  const uint4* Bq1 = reinterpret_cast<const uint4*>(Bm + (size_t)rB1 * K) + c4;

  // this thread's LDS destinations, per buffer
  uint32_t dxA0[2], dxA1[2], dwB0[2], dwB1[2];
#pragma unroll
  for (int b = 0; b < 2; ++b) {
    dxA0[b] = lds_off(&lx[b][lrow0 * TLDS + c4 * 4]);
    dxA1[b] = lds_off(&lx[b][lrow1 * TLDS + c4 * 4]);
    dwB0[b] = lds_off(&lw[b][lrow0 * TLDS + c4 * 4]);
    dwB1[b] = lds_off(&lw[b][lrow1 * TLDS + c4 * 4]);
  }

  v8f acc[2][4];
#pragma unroll
  for (int i = 0; i < 2; ++i)
#pragma unroll
    for (int j = 0; j < 4; ++j)
#pragma unroll
      for (int r = 0; r < 8; ++r) acc[i][j][r] = 0.0f;

  // prologue: stream tile 0 into buffer 0
  async_load_b128(dxA0[0], Aq0);
  async_load_b128(dxA1[0], Aq1);
  async_load_b128(dwB0[0], Bq0);
  async_load_b128(dwB1[0], Bq1);

  const int nsteps = K / 32;           // 32 bf16 = 4 uint4 per step
  for (int s = 0; s < nsteps; ++s) {
    const int buf = s & 1;
    // own async portion of tile s has landed; barrier covers all waves
    wait_asynccnt0();
    __syncthreads();

    // stream tile s+1 into the other buffer (its previous readers finished
    // before the barrier above); latency hidden by this tile's WMMAs
    if (s + 1 < nsteps) {
      const int ko = (s + 1) * 4;
      const int nb = buf ^ 1;
      async_load_b128(dxA0[nb], Aq0 + ko);
      async_load_b128(dxA1[nb], Aq1 + ko);
      async_load_b128(dwB0[nb], Bq0 + ko);
      async_load_b128(dwB1[nb], Bq1 + ko);
    }
    // warm L2 two steps ahead (global_prefetch_b8)
    if (s + 2 < nsteps) {
      const int kp = (s + 2) * 4;
      __builtin_prefetch(Aq0 + kp, 0, 1);
      __builtin_prefetch(Bq0 + kp, 0, 1);
    }

    const unsigned int* px = &lx[buf][0];
    const unsigned int* pw = &lw[buf][0];

    // Fragment gathers per documented 16-bit 16x32 layout:
    // lane<16: K in {0..7, 16..23};  lane>=16: K in {8..15, 24..31}.
    // As packed dwords: q[0] = dwords half*4..+3, q[1] = dwords 8+half*4..+3.
    FragBF a[2], bf[4];
#pragma unroll
    for (int i = 0; i < 2; ++i) {
      const uint4* p = reinterpret_cast<const uint4*>(
          &px[(wm * 32 + i * 16 + rsel) * TLDS]);
      a[i].q[0] = p[half];
      a[i].q[1] = p[2 + half];
    }
#pragma unroll
    for (int j = 0; j < 4; ++j) {
      const uint4* p = reinterpret_cast<const uint4*>(
          &pw[(wn * 64 + j * 16 + rsel) * TLDS]);
      bf[j].q[0] = p[half];
      bf[j].q[1] = p[2 + half];
    }

#pragma unroll
    for (int i = 0; i < 2; ++i)
#pragma unroll
      for (int j = 0; j < 4; ++j)
        acc[i][j] = __builtin_amdgcn_wmma_f32_16x16x32_bf16(
            false, a[i].v, false, bf[j].v, (short)0, acc[i][j], false, false);
    // no trailing barrier: the next iteration's async writes target the other
    // buffer, and the barrier at its top orders them against this buffer
  }

  // Epilogue: C/D layout -> lane: N = rsel, M = vgpr + half*8.
#pragma unroll
  for (int i = 0; i < 2; ++i) {
#pragma unroll
    for (int j = 0; j < 4; ++j) {
      int n = bn + wn * 64 + j * 16 + rsel;
      float bv = bias[n];
#pragma unroll
      for (int r = 0; r < 8; ++r) {
        int m = bm + wm * 32 + i * 16 + r + half * 8;
        if (m < M) store_out(&C[(long long)m * ldc + n], acc[i][j][r] + bv);
      }
    }
  }
}

// ---------------------------------------------------------------------------
// Class-attention: one block per (b,h). q0: [64,768] f32, KV: [36928,1536]
// bf16 (cols 0..767 = K, 768..1535 = V). Writes cls bf16 [64,768].
// ---------------------------------------------------------------------------
__global__ __launch_bounds__(256)
void class_attn_kernel(const float* __restrict__ q0,
                       const unsigned short* __restrict__ KV,
                       unsigned short* __restrict__ cls)
{
  const int bh = blockIdx.x;            // 0..767
  const int b = bh / HEADS, h = bh % HEADS;
  const int tid = threadIdx.x;

  __shared__ float qs[HDIM];
  __shared__ float s[NTOK];
  __shared__ float red[256];
  __shared__ float part[4][HDIM];

  if (tid < HDIM) qs[tid] = q0[b * C_DIM + h * HDIM + tid];
  __syncthreads();

  const size_t rowbase = (size_t)b * NTOK * KV_N + h * HDIM;

  // scores + running max
  float lmax = -3.4e38f;
  for (int n = tid; n < NTOK; n += 256) {
    const unsigned int* kr =
        reinterpret_cast<const unsigned int*>(KV + rowbase + (size_t)n * KV_N);
    float acc = 0.0f;
#pragma unroll
    for (int d2 = 0; d2 < HDIM / 2; ++d2) {
      unsigned int u = kr[d2];
      acc += qs[2 * d2]     * bf2f((unsigned short)(u & 0xFFFFu));
      acc += qs[2 * d2 + 1] * bf2f((unsigned short)(u >> 16));
    }
    acc *= SCALE_F;
    s[n] = acc;
    lmax = fmaxf(lmax, acc);
  }
  red[tid] = lmax;
  __syncthreads();
  for (int st = 128; st >= 1; st >>= 1) {
    if (tid < st) red[tid] = fmaxf(red[tid], red[tid + st]);
    __syncthreads();
  }
  const float smax = red[0];
  __syncthreads();

  // exp + sum
  float lsum = 0.0f;
  for (int n = tid; n < NTOK; n += 256) {
    float p = __expf(s[n] - smax);
    s[n] = p;
    lsum += p;
  }
  red[tid] = lsum;
  __syncthreads();
  for (int st = 128; st >= 1; st >>= 1) {
    if (tid < st) red[tid] += red[tid + st];
    __syncthreads();
  }
  const float inv_sum = 1.0f / red[0];
  __syncthreads();

  // weighted V: thread (j = tid>>6, d = tid&63) accumulates n strided by 4
  const int d = tid & 63, j = tid >> 6;
  const unsigned short* Vb = KV + rowbase + 768 + d;   // V columns start at 768
  float acc = 0.0f;
  for (int n = j; n < NTOK; n += 4)
    acc += s[n] * bf2f(Vb[(size_t)n * KV_N]);
  part[j][d] = acc;
  __syncthreads();

  if (tid < HDIM) {
    float v = (part[0][tid] + part[1][tid] + part[2][tid] + part[3][tid]) * inv_sum;
    cls[b * C_DIM + h * HDIM + tid] = f2bf(v);
  }
}

// ---------------------------------------------------------------------------
// Host launch
// ---------------------------------------------------------------------------
extern "C" void kernel_launch(void* const* d_in, const int* in_sizes, int n_in,
                              void* d_out, int out_size, void* d_ws, size_t ws_size,
                              hipStream_t stream) {
  (void)in_sizes; (void)n_in; (void)out_size; (void)ws_size;

  const float* x      = (const float*)d_in[0];   // [64,577,768]
  const float* qkv_w  = (const float*)d_in[1];   // [2304,768]
  const float* qkv_b  = (const float*)d_in[2];   // [2304]
  const float* proj_w = (const float*)d_in[3];   // [768,768]
  const float* proj_b = (const float*)d_in[4];   // [768]
  float* out = (float*)d_out;                    // [64,577,768]

  // workspace layout (256B aligned), ~175 MB total
  char* ws = (char*)d_ws;
  size_t off = 0;
  auto carve = [&](size_t bytes) {
    char* p = ws + off;
    off = (off + bytes + 255) & ~(size_t)255;
    return p;
  };
  unsigned short* x_bf    = (unsigned short*)carve((size_t)MROWS * C_DIM * 2);
  unsigned short* w_bf    = (unsigned short*)carve((size_t)3 * C_DIM * C_DIM * 2);
  unsigned short* pw_bf   = (unsigned short*)carve((size_t)C_DIM * C_DIM * 2);
  unsigned short* xcls_bf = (unsigned short*)carve((size_t)BATCH * C_DIM * 2);
  unsigned short* kv_bf   = (unsigned short*)carve((size_t)MROWS * KV_N * 2);
  float*          q0f     = (float*)carve((size_t)BATCH * C_DIM * 4);
  unsigned short* cls_bf  = (unsigned short*)carve((size_t)BATCH * C_DIM * 2);

  // 1) precision converts
  {
    int n = MROWS * C_DIM;
    f32_to_bf16_kernel<<<(n + 255) / 256, 256, 0, stream>>>(x, x_bf, n);
  }
  {
    int n = 3 * C_DIM * C_DIM;
    f32_to_bf16_kernel<<<(n + 255) / 256, 256, 0, stream>>>(qkv_w, w_bf, n);
  }
  {
    int n = C_DIM * C_DIM;
    f32_to_bf16_kernel<<<(n + 255) / 256, 256, 0, stream>>>(proj_w, pw_bf, n);
  }
  gather_cls_kernel<<<(BATCH * C_DIM + 255) / 256, 256, 0, stream>>>(x, xcls_bf);

  // 2) passthrough: out[:,1:] = x[:,1:] (copy all; row 0 overwritten by proj)
  hipMemcpyAsync(out, x, (size_t)MROWS * C_DIM * sizeof(float),
                 hipMemcpyDeviceToDevice, stream);

  // 3) K,V GEMM: [36928,768] @ [1536,768]^T + qkv_b[768:] -> bf16 [36928,1536]
  {
    dim3 grid(KV_N / 128, (MROWS + 127) / 128);
    gemm_bf16_nt<unsigned short><<<grid, 256, 0, stream>>>(
        x_bf, w_bf + (size_t)C_DIM * C_DIM, qkv_b + C_DIM,
        kv_bf, (long long)KV_N, MROWS, KV_N, C_DIM);
  }

  // 4) q0 GEMM (CLS rows only): [64,768] @ [768,768]^T + qkv_b[:768] -> f32
  {
    dim3 grid(C_DIM / 128, 1);
    gemm_bf16_nt<float><<<grid, 256, 0, stream>>>(
        xcls_bf, w_bf, qkv_b, q0f, (long long)C_DIM, BATCH, C_DIM, C_DIM);
  }

  // 5) softmax attention per (b,h) -> cls bf16 [64,768]
  class_attn_kernel<<<BATCH * HEADS, 256, 0, stream>>>(q0f, kv_bf, cls_bf);

  // 6) proj GEMM: [64,768] @ [768,768]^T + proj_b, strided into out[b,0,:]
  {
    dim3 grid(C_DIM / 128, 1);
    gemm_bf16_nt<float><<<grid, 256, 0, stream>>>(
        cls_bf, pw_bf, proj_b, out, (long long)NTOK * C_DIM,
        BATCH, C_DIM, C_DIM);
  }
}
